// LiquidEmbedding_64536178589854
// MI455X (gfx1250) — compile-verified, matrix operand order checked by reading, not run on hardware
//
#include <hip/hip_runtime.h>

// ---------------------------------------------------------------------------
// LiquidEmbedding forward for MI455X (gfx1250, wave32, WMMA).
// Heavy math (2 convs-as-GEMM K=3072, proj, 2 ph projections) runs through
// v_wmma_f32_16x16x32_bf16 with LDS-staged, async-copied, double-buffered B
// tiles and register-pipelined A fragments. FFT/LN/loss are VALU/LDS kernels.
// Workspace usage: ~353 MB (see offsets below).
// ---------------------------------------------------------------------------

#define B_  8
#define S_  2048
#define D_  1024
#define SP_ 2050          // S + 2 zero-pad rows for conv windows
#define M_  (B_ * S_)     // 16384

typedef __attribute__((ext_vector_type(16))) __bf16 bf16x16;
typedef __attribute__((ext_vector_type(8)))  __bf16 bf16x8;
typedef __attribute__((ext_vector_type(8)))  float  f32x8;
typedef __attribute__((__vector_size__(16))) int    vi4;   // matches builtin param

union FragAB { bf16x16 v; bf16x8 h[2]; };

#define AS1 __attribute__((address_space(1)))
#define AS3 __attribute__((address_space(3)))

// Async global->LDS copy of 16 bytes (one b128 transfer per lane).
// Falls back to a synchronous copy if the builtin is unavailable.
static __device__ __forceinline__ void asyncCopy16(const __bf16* g, void* l) {
#if __has_builtin(__builtin_amdgcn_global_load_async_to_lds_b128)
  __builtin_amdgcn_global_load_async_to_lds_b128(
      (AS1 vi4*)g, (AS3 vi4*)l, 0, 0);
#else
  *(uint4*)l = *(const uint4*)g;
#endif
}

static __device__ __forceinline__ void asyncWait2() {
#if __has_builtin(__builtin_amdgcn_s_wait_asynccnt)
  __builtin_amdgcn_s_wait_asynccnt(2);   // allow the 2 just-issued fills to fly
#endif
}

static __device__ __forceinline__ __bf16 f2bf(float x) {
  union { float f; unsigned u; } in; in.f = x;
  unsigned r = in.u + 0x7FFFu + ((in.u >> 16) & 1u);   // round-to-nearest-even
  union { unsigned short s; __bf16 b; } out; out.s = (unsigned short)(r >> 16);
  return out.b;
}

static __device__ __forceinline__ float gelu_exact(float v) {
  return 0.5f * v * (1.0f + erff(v * 0.70710678118654752f));
}

static __device__ __forceinline__ float blockReduceSum256(float v, float* red) {
  int tid = threadIdx.x;
  red[tid] = v; __syncthreads();
  #pragma unroll
  for (int s = 128; s > 0; s >>= 1) {
    if (tid < s) red[tid] += red[tid + s];
    __syncthreads();
  }
  float r = red[0]; __syncthreads();
  return r;
}

static __device__ __forceinline__ float blockReduceMax256(float v, float* red) {
  int tid = threadIdx.x;
  red[tid] = v; __syncthreads();
  #pragma unroll
  for (int s = 128; s > 0; s >>= 1) {
    if (tid < s) red[tid] = fmaxf(red[tid], red[tid + s]);
    __syncthreads();
  }
  float r = red[0]; __syncthreads();
  return r;
}

// ---------------------------------------------------------------------------
// 1) Embedding gather + positional add -> zero-padded (B, SP, D) bf16
// ---------------------------------------------------------------------------
__global__ void le_embed_pad(const int* __restrict__ x,
                             const float* __restrict__ tok,
                             const float* __restrict__ pos,
                             __bf16* __restrict__ out) {
  size_t idx = (size_t)blockIdx.x * 256 + threadIdx.x;
  if (idx >= (size_t)B_ * SP_ * D_) return;
  int c = (int)(idx % D_);
  size_t t = idx / D_;
  int sp = (int)(t % SP_);
  int b  = (int)(t / SP_);
  if (sp == 0 || sp == SP_ - 1) { out[idx] = f2bf(0.0f); return; }
  int s = sp - 1;
  int tk = x[b * S_ + s];
  float v = tok[(size_t)tk * D_ + c] + pos[(size_t)s * D_ + c];
  out[idx] = f2bf(v);
}

// ---------------------------------------------------------------------------
// 2) Conv weight permute (co, ci, k) -> Bt[co][k*D + ci] bf16
// ---------------------------------------------------------------------------
__global__ void le_prep_conv_wt(const float* __restrict__ w, __bf16* __restrict__ wt) {
  int idx = blockIdx.x * 256 + threadIdx.x;          // over D_ * 3*D_
  if (idx >= D_ * 3 * D_) return;
  int co = idx / (3 * D_);
  int j  = idx % (3 * D_);
  int k  = j / D_;
  int ci = j % D_;
  wt[idx] = f2bf(w[((size_t)co * D_ + ci) * 3 + k]);
}

__global__ void le_cast_bf16(const float* __restrict__ in, __bf16* __restrict__ out, int n) {
  int i = blockIdx.x * 256 + threadIdx.x;
  if (i < n) out[i] = f2bf(in[i]);
}

// ---------------------------------------------------------------------------
// 3) WMMA bf16 GEMM:  C[m][n] = sum_k A[m][k] * Bt[n][k] + bias[n]
//    A rows may overlap (conv windows):  row offset = (m + (m/rowsPerBatch)*batchSkip)*lda
//    Block = 128 threads = 4 waves; block tile 128x64.
//    Each wave owns a 32x64 tile: 2 A fragments x 4 B fragments = 8 accumulators.
//    B tile (shared by all 4 waves) is staged in LDS via async b128 copies,
//    double-buffered; A fragments are register-pipelined one k-step ahead.
//    LDS B rows padded to 80 B -> conflict-free ds_load_b128.
// ---------------------------------------------------------------------------
#define BSTRIDE 80

__global__ __launch_bounds__(128)
void le_wmma_gemm_bf16(const __bf16* __restrict__ A, const __bf16* __restrict__ Bt,
                       float* __restrict__ C, const float* __restrict__ bias,
                       int N, int K, int lda, int rowsPerBatch, int batchSkip) {
  __shared__ __align__(16) unsigned char bstage[2][64 * BSTRIDE];

  const int tid  = threadIdx.x;
  const int lane = tid & 31;
  const int wave = tid >> 5;
  const int lrow = lane & 15;
  const int khalf = (lane >> 4) << 3;          // element offset 0 / 8 (lane half)
  const int kbyte = khalf << 1;                // byte offset in an LDS B row
  const int ncol  = blockIdx.x * 64;
  const int mbase = blockIdx.y * 128 + wave * 32;

  const int m0 = mbase + lrow;
  const int m1 = mbase + 16 + lrow;
  const __bf16* __restrict__ arow0 =
      A + ((size_t)m0 + (size_t)(m0 / rowsPerBatch) * batchSkip) * (size_t)lda;
  const __bf16* __restrict__ arow1 =
      A + ((size_t)m1 + (size_t)(m1 / rowsPerBatch) * batchSkip) * (size_t)lda;

  // B staging: thread copies 32 B (two b128s) of tile row n = tid>>1, half tid&1
  const int bn = tid >> 1, bh = tid & 1;
  const __bf16* __restrict__ gB = Bt + (size_t)(ncol + bn) * (size_t)K + bh * 16;
  unsigned char* lB0 = &bstage[0][bn * BSTRIDE + bh * 32];
  unsigned char* lB1 = &bstage[1][bn * BSTRIDE + bh * 32];

  f32x8 acc[2][4];
  #pragma unroll
  for (int i = 0; i < 2; ++i)
    #pragma unroll
    for (int j = 0; j < 4; ++j)
      #pragma unroll
      for (int r = 0; r < 8; ++r) acc[i][j][r] = 0.0f;

  const int nsteps = K >> 5;

  // Prologue: fill stage 0 (async), prefetch A fragments for step 0
  asyncCopy16(gB,     lB0);
  asyncCopy16(gB + 8, lB0 + 16);
  FragAB a0c, a1c;
  a0c.h[0] = *reinterpret_cast<const bf16x8*>(arow0 + khalf);
  a0c.h[1] = *reinterpret_cast<const bf16x8*>(arow0 + khalf + 16);
  a1c.h[0] = *reinterpret_cast<const bf16x8*>(arow1 + khalf);
  a1c.h[1] = *reinterpret_cast<const bf16x8*>(arow1 + khalf + 16);
  FragAB a0n = a0c, a1n = a1c;

  for (int s = 0; s < nsteps; ++s) {
    const int kb = s << 5;
    const bool more = (s + 1 < nsteps);
    if (more) {
      // Fill the other stage (its consumers finished at the end-barrier of s-1)
      unsigned char* lB = (s & 1) ? lB0 : lB1;
      asyncCopy16(gB + kb + 32, lB);
      asyncCopy16(gB + kb + 40, lB + 16);
      // Register-pipeline next A fragments
      a0n.h[0] = *reinterpret_cast<const bf16x8*>(arow0 + kb + 32 + khalf);
      a0n.h[1] = *reinterpret_cast<const bf16x8*>(arow0 + kb + 48 + khalf);
      a1n.h[0] = *reinterpret_cast<const bf16x8*>(arow1 + kb + 32 + khalf);
      a1n.h[1] = *reinterpret_cast<const bf16x8*>(arow1 + kb + 48 + khalf);
    }
    asyncWait2();          // this wave's stage-s fills landed
    __syncthreads();       // all waves' fills landed

    const unsigned char* sb = bstage[s & 1];
    #pragma unroll
    for (int j = 0; j < 4; ++j) {
      FragAB bfr;
      const unsigned char* bp = sb + (j * 16 + lrow) * BSTRIDE + kbyte;
      bfr.h[0] = *reinterpret_cast<const bf16x8*>(bp);
      bfr.h[1] = *reinterpret_cast<const bf16x8*>(bp + 32);
      acc[0][j] = __builtin_amdgcn_wmma_f32_16x16x32_bf16(
          false, a0c.v, false, bfr.v, (short)0, acc[0][j], false, false);
      acc[1][j] = __builtin_amdgcn_wmma_f32_16x16x32_bf16(
          false, a1c.v, false, bfr.v, (short)0, acc[1][j], false, false);
    }
    __syncthreads();       // stage s fully consumed; safe to refill next iter
    a0c = a0n; a1c = a1n;
  }

  const int rtop = (lane >> 4) << 3;   // +8 for high lanes
  #pragma unroll
  for (int i = 0; i < 2; ++i) {
    const int rbase = mbase + i * 16 + rtop;
    #pragma unroll
    for (int j = 0; j < 4; ++j) {
      const int n = ncol + j * 16 + lrow;
      const float bv = bias[n];
      #pragma unroll
      for (int r = 0; r < 8; ++r) {
        C[(size_t)(rbase + r) * N + n] = acc[i][j][r] + bv;
      }
    }
  }
}

// ---------------------------------------------------------------------------
// 4) GELU + re-pad to (B, SP, D) bf16 (input to conv2)
// ---------------------------------------------------------------------------
__global__ void le_gelu_pad(const float* __restrict__ in, __bf16* __restrict__ out) {
  size_t idx = (size_t)blockIdx.x * 256 + threadIdx.x;
  if (idx >= (size_t)B_ * SP_ * D_) return;
  int c = (int)(idx % D_);
  size_t t = idx / D_;
  int sp = (int)(t % SP_);
  int b  = (int)(t / SP_);
  if (sp == 0 || sp == SP_ - 1) { out[idx] = f2bf(0.0f); return; }
  float v = in[((size_t)(b * S_ + sp - 1)) * D_ + c];
  out[idx] = f2bf(gelu_exact(v));
}

// ---------------------------------------------------------------------------
// 5) GELU + LayerNorm (conv2 raw -> xe f32), one block per row of 1024
// ---------------------------------------------------------------------------
__global__ __launch_bounds__(256)
void le_gelu_ln(const float* __restrict__ in, const float* __restrict__ g,
                const float* __restrict__ bln, float* __restrict__ out) {
  __shared__ float red[256];
  int m = blockIdx.x, tid = threadIdx.x;
  const float* row = in + (size_t)m * D_;
  float vals[4]; float ls = 0.0f;
  #pragma unroll
  for (int i = 0; i < 4; ++i) {
    float v = gelu_exact(row[tid + 256 * i]);
    vals[i] = v; ls += v;
  }
  float mu = blockReduceSum256(ls, red) * (1.0f / D_);
  float lv = 0.0f;
  #pragma unroll
  for (int i = 0; i < 4; ++i) { float d = vals[i] - mu; lv += d * d; }
  float var = blockReduceSum256(lv, red) * (1.0f / D_);
  float rstd = rsqrtf(var + 1e-5f);
  #pragma unroll
  for (int i = 0; i < 4; ++i) {
    int c = tid + 256 * i;
    out[(size_t)m * D_ + c] = (vals[i] - mu) * rstd * g[c] + bln[c];
  }
}

// ---------------------------------------------------------------------------
// 6) LayerNorm (proj_real -> x_out f32 to d_out, plus bf16 copy)
// ---------------------------------------------------------------------------
__global__ __launch_bounds__(256)
void le_ln_out(const float* __restrict__ in, const float* __restrict__ g,
               const float* __restrict__ bln, float* __restrict__ outf,
               __bf16* __restrict__ outb) {
  __shared__ float red[256];
  int m = blockIdx.x, tid = threadIdx.x;
  const float* row = in + (size_t)m * D_;
  float vals[4]; float ls = 0.0f;
  #pragma unroll
  for (int i = 0; i < 4; ++i) { vals[i] = row[tid + 256 * i]; ls += vals[i]; }
  float mu = blockReduceSum256(ls, red) * (1.0f / D_);
  float lv = 0.0f;
  #pragma unroll
  for (int i = 0; i < 4; ++i) { float d = vals[i] - mu; lv += d * d; }
  float var = blockReduceSum256(lv, red) * (1.0f / D_);
  float rstd = rsqrtf(var + 1e-5f);
  #pragma unroll
  for (int i = 0; i < 4; ++i) {
    int c = tid + 256 * i;
    float y = (vals[i] - mu) * rstd * g[c] + bln[c];
    outf[(size_t)m * D_ + c] = y;
    outb[(size_t)m * D_ + c] = f2bf(y);
  }
}

// ---------------------------------------------------------------------------
// 7) Forward FFT (2048-pt radix-2 DIT in LDS) per (batch, channel) column,
//    fused spectral-complexity count (strict >, 0.1*max threshold).
// ---------------------------------------------------------------------------
__global__ __launch_bounds__(256)
void le_fft_fwd(const float* __restrict__ xe, float* __restrict__ fre,
                float* __restrict__ fim, float* __restrict__ counts) {
  __shared__ float sre[2048];
  __shared__ float sim[2048];
  __shared__ float red[256];
  int blk = blockIdx.x;
  int b  = blk >> 10;
  int ch = blk & (D_ - 1);
  int tid = threadIdx.x;
  const float* col = xe + (size_t)b * S_ * D_ + ch;

  for (int i = tid; i < 2048; i += 256) {
    int r = (int)(__brev((unsigned)i) >> 21);     // 11-bit reversal
    sre[r] = col[(size_t)i * D_];
    sim[r] = 0.0f;
  }
  __syncthreads();

  for (int len = 2; len <= 2048; len <<= 1) {
    int half = len >> 1;
    float ang0 = -6.28318530717958648f / (float)len;
    for (int j = tid; j < 1024; j += 256) {
      int pos = j & (half - 1);
      int i0 = ((j - pos) << 1) + pos;
      int i1 = i0 + half;
      float ang = ang0 * (float)pos;
      float c = __cosf(ang), s = __sinf(ang);
      float xr = sre[i1], xi = sim[i1];
      float tr = c * xr - s * xi;
      float ti = c * xi + s * xr;
      float ur = sre[i0], ui = sim[i0];
      sre[i0] = ur + tr; sim[i0] = ui + ti;
      sre[i1] = ur - tr; sim[i1] = ui - ti;
    }
    __syncthreads();
  }

  // store spectrum + complexity count (on squared magnitudes; equivalent)
  float lmax = 0.0f;
  size_t base = (size_t)b * S_ * D_ + ch;
  for (int f = tid; f < 2048; f += 256) {
    float re = sre[f], im = sim[f];
    fre[base + (size_t)f * D_] = re;
    fim[base + (size_t)f * D_] = im;
    lmax = fmaxf(lmax, re * re + im * im);
  }
  float mx2 = blockReduceMax256(lmax, red);
  float th2 = 0.01f * mx2;
  float lcnt = 0.0f;
  for (int f = tid; f < 2048; f += 256) {
    float re = sre[f], im = sim[f];
    if (re * re + im * im > th2) lcnt += 1.0f;
  }
  float cnt = blockReduceSum256(lcnt, red);
  if (tid == 0) atomicAdd(&counts[b], cnt);
}

__global__ void le_zero_counts(float* counts) {
  if (threadIdx.x < B_) counts[threadIdx.x] = 0.0f;
}

__global__ void le_compute_nkeep(const float* __restrict__ counts, int* __restrict__ nkeep) {
  int b = threadIdx.x;
  if (b < B_) {
    float cx = counts[b] * (1.0f / ((float)S_ * (float)D_));
    float ratio = 0.5f * (1.0f - cx);
    ratio = fminf(fmaxf(ratio, 0.1f), 1.0f);
    int n = (int)(ratio * (float)S_);      // trunc, matches astype(int32)
    n = max(1, min(n, S_));
    nkeep[b] = n;
  }
}

// ---------------------------------------------------------------------------
// 8) Masked inverse FFT -> real part, written directly as bf16 GEMM operand
// ---------------------------------------------------------------------------
__global__ __launch_bounds__(256)
void le_ifft_masked(const float* __restrict__ fre, const float* __restrict__ fim,
                    const int* __restrict__ nkeep, __bf16* __restrict__ realb) {
  __shared__ float sre[2048];
  __shared__ float sim[2048];
  int blk = blockIdx.x;
  int b  = blk >> 10;
  int ch = blk & (D_ - 1);
  int tid = threadIdx.x;
  int N = nkeep[b];
  size_t base = (size_t)b * S_ * D_ + ch;

  for (int i = tid; i < 2048; i += 256) {
    int r = (int)(__brev((unsigned)i) >> 21);
    float vr = 0.0f, vi = 0.0f;
    if (i < N) {
      vr = fre[base + (size_t)i * D_];
      vi = fim[base + (size_t)i * D_];
    }
    sre[r] = vr; sim[r] = vi;
  }
  __syncthreads();

  for (int len = 2; len <= 2048; len <<= 1) {
    int half = len >> 1;
    float ang0 = 6.28318530717958648f / (float)len;   // inverse sign
    for (int j = tid; j < 1024; j += 256) {
      int pos = j & (half - 1);
      int i0 = ((j - pos) << 1) + pos;
      int i1 = i0 + half;
      float ang = ang0 * (float)pos;
      float c = __cosf(ang), s = __sinf(ang);
      float xr = sre[i1], xi = sim[i1];
      float tr = c * xr - s * xi;
      float ti = c * xi + s * xr;
      float ur = sre[i0], ui = sim[i0];
      sre[i0] = ur + tr; sim[i0] = ui + ti;
      sre[i1] = ur - tr; sim[i1] = ui - ti;
    }
    __syncthreads();
  }

  const float scale = 1.0f / 2048.0f;
  for (int sIdx = tid; sIdx < 2048; sIdx += 256) {
    realb[((size_t)(b * S_ + sIdx)) * D_ + ch] = f2bf(sre[sIdx] * scale);
  }
}

// ---------------------------------------------------------------------------
// 9) Per-row cosine / smooth-L1 stats
// ---------------------------------------------------------------------------
__global__ __launch_bounds__(256)
void le_row_cos_l1(const float* __restrict__ xo, const float* __restrict__ pr,
                   float* __restrict__ rc, float* __restrict__ rl1) {
  __shared__ float red[256];
  int m = blockIdx.x, tid = threadIdx.x;
  const float* xrow = xo + (size_t)m * D_;
  const float* trow = pr + (size_t)m * D_;
  float sxx = 0, stt = 0, sxt = 0, sl1 = 0;
  for (int c = tid; c < D_; c += 256) {
    float x = xrow[c], t = trow[c];
    sxx += x * x; stt += t * t; sxt += x * t;
    float d = x - t, ad = fabsf(d);
    sl1 += (ad < 1.0f) ? 0.5f * d * d : ad - 0.5f;
  }
  sxx = blockReduceSum256(sxx, red);
  stt = blockReduceSum256(stt, red);
  sxt = blockReduceSum256(sxt, red);
  sl1 = blockReduceSum256(sl1, red);
  if (tid == 0) {
    float nx = fmaxf(sqrtf(sxx), 1e-12f);
    float nt = fmaxf(sqrtf(stt), 1e-12f);
    float cs = sxt / (nx * nt);
    cs = fminf(fmaxf(cs, -1.0f + 1e-8f), 1.0f - 1e-8f);
    rc[m]  = 1.0f - cs;
    rl1[m] = sl1 * (1.0f / D_);
  }
}

// ---------------------------------------------------------------------------
// 10) Per-row semantic term: mean((l2n(px) - l2n(pt))^2)
// ---------------------------------------------------------------------------
__global__ __launch_bounds__(256)
void le_row_sem(const float* __restrict__ px, const float* __restrict__ pt,
                float* __restrict__ rsem) {
  __shared__ float red[256];
  int m = blockIdx.x, tid = threadIdx.x;
  const float* xr = px + (size_t)m * D_;
  const float* tr = pt + (size_t)m * D_;
  float sxx = 0, stt = 0;
  for (int c = tid; c < D_; c += 256) {
    float x = xr[c], t = tr[c];
    sxx += x * x; stt += t * t;
  }
  sxx = blockReduceSum256(sxx, red);
  stt = blockReduceSum256(stt, red);
  float inx = 1.0f / fmaxf(sqrtf(sxx), 1e-12f);
  float int_ = 1.0f / fmaxf(sqrtf(stt), 1e-12f);
  float sd = 0;
  for (int c = tid; c < D_; c += 256) {
    float d = xr[c] * inx - tr[c] * int_;
    sd += d * d;
  }
  sd = blockReduceSum256(sd, red);
  if (tid == 0) rsem[m] = sd * (1.0f / D_);
}

// ---------------------------------------------------------------------------
// 11) Final scalar:  loss = (0.6*meanAll(1-cos)*maskSum + sum_mask(0.2*l1+0.2*sem))
//                           / (maskSum + 1e-8)
// ---------------------------------------------------------------------------
__global__ __launch_bounds__(256)
void le_final_loss(const float* __restrict__ rc, const float* __restrict__ rl1,
                   const float* __restrict__ rsem, const int* __restrict__ nkeep,
                   float* __restrict__ out_loss) {
  __shared__ float redA[256];
  __shared__ float redB[256];
  int tid = threadIdx.x;
  float sc = 0.0f, sm = 0.0f;
  for (int m = tid; m < M_; m += 256) {
    sc += rc[m];
    int b = m >> 11;
    int s = m & (S_ - 1);
    if (s < nkeep[b]) sm += 0.2f * rl1[m] + 0.2f * rsem[m];
  }
  redA[tid] = sc; redB[tid] = sm; __syncthreads();
  #pragma unroll
  for (int st = 128; st > 0; st >>= 1) {
    if (tid < st) { redA[tid] += redA[tid + st]; redB[tid] += redB[tid + st]; }
    __syncthreads();
  }
  if (tid == 0) {
    float masksum = 0.0f;
    for (int b = 0; b < B_; ++b) masksum += (float)nkeep[b];
    float meanc = redA[0] * (1.0f / (float)M_);
    out_loss[0] = (0.6f * meanc * masksum + redB[0]) / (masksum + 1e-8f);
  }
}

// ---------------------------------------------------------------------------
// Workspace layout (bytes)
// ---------------------------------------------------------------------------
#define WS_PADA   ((size_t)0)            // 33,587,200  emb pad bf16 -> real bf16 -> x_out bf16
#define WS_PADB   ((size_t)33587200)     // 33,587,200  h1 pad bf16 -> proj_real bf16
#define WS_W1T    ((size_t)67174400)     //  6,291,456
#define WS_W2T    ((size_t)73465856)     //  6,291,456
#define WS_WPT    ((size_t)79757312)     //  2,097,152
#define WS_WHT    ((size_t)81854464)     //  2,097,152
#define WS_C      ((size_t)83951616)     // 67,108,864  conv1 raw -> conv2 raw -> proj_real f32
#define WS_X      ((size_t)151060480)    // 67,108,864  xe f32
#define WS_FRE    ((size_t)218169344)    // 67,108,864  fft re -> phx f32
#define WS_FIM    ((size_t)285278208)    // 67,108,864  fft im -> pht f32
#define WS_RC     ((size_t)352387072)    //     65,536
#define WS_RL1    ((size_t)352452608)    //     65,536
#define WS_RSEM   ((size_t)352518144)    //     65,536
#define WS_CNT    ((size_t)352583680)    //         32
#define WS_NKEEP  ((size_t)352583712)    //         32

extern "C" void kernel_launch(void* const* d_in, const int* in_sizes, int n_in,
                              void* d_out, int out_size, void* d_ws, size_t ws_size,
                              hipStream_t stream) {
  (void)in_sizes; (void)n_in; (void)out_size; (void)ws_size;

  const int*   x    = (const int*)  d_in[0];
  const float* tok  = (const float*)d_in[1];
  const float* pos  = (const float*)d_in[2];
  const float* c1w  = (const float*)d_in[3];
  const float* c1b  = (const float*)d_in[4];
  const float* c2w  = (const float*)d_in[5];
  const float* c2b  = (const float*)d_in[6];
  const float* lng  = (const float*)d_in[7];
  const float* lnb  = (const float*)d_in[8];
  const float* pjw  = (const float*)d_in[9];
  const float* pjb  = (const float*)d_in[10];
  const float* phw  = (const float*)d_in[11];
  const float* phb  = (const float*)d_in[12];

  char* ws = (char*)d_ws;
  __bf16* padA  = (__bf16*)(ws + WS_PADA);
  __bf16* padB  = (__bf16*)(ws + WS_PADB);
  __bf16* w1t   = (__bf16*)(ws + WS_W1T);
  __bf16* w2t   = (__bf16*)(ws + WS_W2T);
  __bf16* wpt   = (__bf16*)(ws + WS_WPT);
  __bf16* wht   = (__bf16*)(ws + WS_WHT);
  float*  bufC  = (float*)(ws + WS_C);
  float*  bufX  = (float*)(ws + WS_X);
  float*  fre   = (float*)(ws + WS_FRE);
  float*  fim   = (float*)(ws + WS_FIM);
  float*  rc    = (float*)(ws + WS_RC);
  float*  rl1   = (float*)(ws + WS_RL1);
  float*  rsem  = (float*)(ws + WS_RSEM);
  float*  cnts  = (float*)(ws + WS_CNT);
  int*    nkeep = (int*)  (ws + WS_NKEEP);

  float* xout = (float*)d_out;                 // M_*D_ f32
  float* loss = (float*)d_out + (size_t)M_ * D_;

  const int padTotalBlocks = (B_ * SP_ * D_ + 255) / 256;   // 65600
  dim3 gemmGrid(D_ / 64, M_ / 128);                         // (16, 128)

  // Stage 1: embedding + weight prep
  le_embed_pad   <<<padTotalBlocks, 256, 0, stream>>>(x, tok, pos, padA);
  le_prep_conv_wt<<<(D_ * 3 * D_) / 256, 256, 0, stream>>>(c1w, w1t);
  le_prep_conv_wt<<<(D_ * 3 * D_) / 256, 256, 0, stream>>>(c2w, w2t);
  le_cast_bf16   <<<(D_ * D_) / 256, 256, 0, stream>>>(pjw, wpt, D_ * D_);
  le_cast_bf16   <<<(D_ * D_) / 256, 256, 0, stream>>>(phw, wht, D_ * D_);

  // Stage 2: conv1 (implicit GEMM, K=3072) -> GELU -> pad -> conv2 -> GELU+LN
  le_wmma_gemm_bf16<<<gemmGrid, 128, 0, stream>>>(padA, w1t, bufC, c1b,
                                                  D_, 3 * D_, D_, S_, 2);
  le_gelu_pad<<<padTotalBlocks, 256, 0, stream>>>(bufC, padB);
  le_wmma_gemm_bf16<<<gemmGrid, 128, 0, stream>>>(padB, w2t, bufC, c2b,
                                                  D_, 3 * D_, D_, S_, 2);
  le_gelu_ln<<<M_, 256, 0, stream>>>(bufC, lng, lnb, bufX);

  // Stage 3: FFT + complexity -> keep counts -> masked iFFT (real, bf16)
  le_zero_counts<<<1, 32, 0, stream>>>(cnts);
  le_fft_fwd<<<B_ * D_, 256, 0, stream>>>(bufX, fre, fim, cnts);
  le_compute_nkeep<<<1, 32, 0, stream>>>(cnts, nkeep);
  le_ifft_masked<<<B_ * D_, 256, 0, stream>>>(fre, fim, nkeep, padA);

  // Stage 4: proj GEMM -> proj_real (f32 + bf16) -> LayerNorm -> x_out
  le_wmma_gemm_bf16<<<gemmGrid, 128, 0, stream>>>(padA, wpt, bufC, pjb,
                                                  D_, D_, D_, M_, 0);
  le_cast_bf16<<<(M_ * D_) / 256, 256, 0, stream>>>(bufC, padB, M_ * D_);
  le_ln_out<<<M_, 256, 0, stream>>>(bufC, lng, lnb, xout, padA);

  // Stage 5: loss — row stats + two ph GEMMs + sem + final reduce
  le_row_cos_l1<<<M_, 256, 0, stream>>>(xout, bufC, rc, rl1);
  le_wmma_gemm_bf16<<<gemmGrid, 128, 0, stream>>>(padA, wht, fre, phb,
                                                  D_, D_, D_, M_, 0);   // proj_x
  le_wmma_gemm_bf16<<<gemmGrid, 128, 0, stream>>>(padB, wht, fim, phb,
                                                  D_, D_, D_, M_, 0);   // proj_t
  le_row_sem<<<M_, 256, 0, stream>>>(fre, fim, rsem);
  le_final_loss<<<1, 256, 0, stream>>>(rc, rl1, rsem, nkeep, loss);
}
